// LocalAttention_10144712753204
// MI455X (gfx1250) — matrix operand order repaired and checked by reading, not compiled, not run on hardware
//
#include <hip/hip_runtime.h>

// ---------------------------------------------------------------------------
// MI455X (gfx1250) windowed local attention, bug-compatible with the JAX ref.
// Path: f32 -> bf16 WMMA (v_wmma_f32_16x16x32_bf16) with f32 accumulation,
// async global->LDS staging (ASYNCcnt) with double buffering in the GEMM.
// ---------------------------------------------------------------------------

typedef __bf16 bf16_t;
typedef __attribute__((ext_vector_type(16))) __bf16 v16bf;
typedef __attribute__((ext_vector_type(8)))  __bf16 v8bf;
typedef __attribute__((ext_vector_type(4)))  __bf16 v4bf;
typedef __attribute__((ext_vector_type(8)))  float  v8f;
typedef __attribute__((ext_vector_type(4)))  int    i32x4;

static __device__ inline v8f zero8() {
    v8f z = {0.f, 0.f, 0.f, 0.f, 0.f, 0.f, 0.f, 0.f};
    return z;
}

#define WMMA_BF16(a, b, c) \
    __builtin_amdgcn_wmma_f32_16x16x32_bf16(false, (a), false, (b), (short)0, (c), false, false)

#if __has_builtin(__builtin_amdgcn_global_load_async_to_lds_b128)
#define ASYNC_LDS 1
#else
#define ASYNC_LDS 0
#endif

// Copy 16 bytes global -> LDS. Async (no VGPR round-trip) when available.
// Builtin signature (from clang diagnostic): (AS1 i32x4*, AS3 i32x4*, imm, imm).
static __device__ inline void async_copy16(const bf16_t* g, bf16_t* l) {
#if ASYNC_LDS
    __builtin_amdgcn_global_load_async_to_lds_b128(
        (__attribute__((address_space(1))) i32x4*)(void*)const_cast<bf16_t*>(g),
        (__attribute__((address_space(3))) i32x4*)(void*)l, 0, 0);
#else
    *(float4*)l = *(const float4*)g;
#endif
}

static __device__ inline void wait_async() {
#if ASYNC_LDS
#if __has_builtin(__builtin_amdgcn_s_wait_asynccnt)
    __builtin_amdgcn_s_wait_asynccnt(0);
#else
    asm volatile("s_wait_asynccnt 0" ::: "memory");
#endif
#endif
}

// Problem constants
#define NB    2
#define SEQ   2048
#define EMB   1024
#define NH    16
#define DH    64
#define WSZ   128
#define MTOT  (NB * SEQ)   // 4096 rows of x

// ===========================================================================
// Kernel 0: f32 -> bf16 elementwise convert (x and the W matrices).
// ===========================================================================
__global__ __launch_bounds__(256) void cvt_kernel(const float* __restrict__ src,
                                                  bf16_t* __restrict__ dst, int n4)
{
    int i = blockIdx.x * 256 + threadIdx.x;
    if (i < n4) {
        float4 f = ((const float4*)src)[i];
        v4bf o = { (bf16_t)f.x, (bf16_t)f.y, (bf16_t)f.z, (bf16_t)f.w };
        ((v4bf*)dst)[i] = o;
    }
}

// ===========================================================================
// Kernel 1: Y = x @ W^T + b  (M=4096, N=1024, K=1024), bf16 in/out, f32 accum.
// 256 threads = 8 waves, 128x128 block tile, TK=64 through double-buffered
// LDS filled with async global->LDS b128 copies. Each wave owns 32(M)x64(N)
// = 2x4 WMMA tiles; 16 WMMAs per K-step.
// ===========================================================================
__global__ __launch_bounds__(256) void qkv_gemm_kernel(
    const bf16_t* __restrict__ xb,
    const bf16_t* __restrict__ Wqb, const bf16_t* __restrict__ Wkb, const bf16_t* __restrict__ Wvb,
    const float* __restrict__ bq, const float* __restrict__ bk, const float* __restrict__ bv,
    bf16_t* __restrict__ Yq, bf16_t* __restrict__ Yk, bf16_t* __restrict__ Yv)
{
    constexpr int TM = 128, TN = 128, TK = 64, LDT = TK + 8;   // 144B row stride (16B aligned)
    __shared__ bf16_t As[2][TM][LDT];
    __shared__ bf16_t Bs[2][TN][LDT];

    const int z = blockIdx.z;
    const bf16_t* Wm   = (z == 0) ? Wqb : (z == 1) ? Wkb : Wvb;
    const float*  bias = (z == 0) ? bq  : (z == 1) ? bk  : bv;
    bf16_t*       Y    = (z == 0) ? Yq  : (z == 1) ? Yk  : Yv;

    const int m0   = blockIdx.x * TM;
    const int n0   = blockIdx.y * TN;
    const int tid  = threadIdx.x;
    const int lane = tid & 31, wave = tid >> 5;
    const int m0w  = (wave >> 1) * 32;   // 4 waves along M
    const int n0w  = (wave & 1) * 64;    // 2 waves along N
    const int lm   = lane & 15, lh = lane >> 4;

    v8f acc[2][4];
#pragma unroll
    for (int mt = 0; mt < 2; ++mt)
#pragma unroll
        for (int nt = 0; nt < 4; ++nt) acc[mt][nt] = zero8();

    // Stage one TK-slab (128 rows x 64 cols bf16 = 8 x 16B per row) of A and B.
    auto stage = [&](int buf, int k0) {
#pragma unroll
        for (int r = 0; r < 4; ++r) {
            int idx = tid + 256 * r;          // 1024 chunks per matrix
            int row = idx >> 3;
            int c16 = (idx & 7) * 8;          // bf16 column offset, 16B chunks
            async_copy16(xb + (size_t)(m0 + row) * EMB + k0 + c16, &As[buf][row][c16]);
            async_copy16(Wm + (size_t)(n0 + row) * EMB + k0 + c16, &Bs[buf][row][c16]);
        }
    };

    constexpr int NK = EMB / TK;   // 16
    stage(0, 0);

    for (int kc = 0; kc < NK; ++kc) {
        wait_async();
        __syncthreads();                       // staged slab visible to all waves
        if (kc + 1 < NK) stage((kc + 1) & 1, (kc + 1) * TK);
        const int bsel = kc & 1;

#pragma unroll
        for (int s = 0; s < 2; ++s) {          // two K=32 halves of the slab
            // A fragments (contiguous k order: element e -> k=16*lh+e)
            v16bf afr[2];
#pragma unroll
            for (int mt = 0; mt < 2; ++mt) {
                const bf16_t* p = &As[bsel][m0w + 16 * mt + lm][32 * s + 16 * lh];
                v16bf a;
#pragma unroll
                for (int e = 0; e < 8; ++e) { a[e] = p[e]; a[8 + e] = p[8 + e]; }
                afr[mt] = a;
            }
            // B fragments (lane holds column n=lm; half lh selects k range 16*lh..+15)
            v16bf bfr[4];
#pragma unroll
            for (int nt = 0; nt < 4; ++nt) {
                const bf16_t* p = &Bs[bsel][n0w + 16 * nt + lm][32 * s + 16 * lh];
                v16bf bb;
#pragma unroll
                for (int e = 0; e < 16; ++e) bb[e] = p[e];
                bfr[nt] = bb;
            }
#pragma unroll
            for (int mt = 0; mt < 2; ++mt)
#pragma unroll
                for (int nt = 0; nt < 4; ++nt)
                    acc[mt][nt] = WMMA_BF16(afr[mt], bfr[nt], acc[mt][nt]);
        }
    }

    // Epilogue: + bias, convert bf16, store. C layout: VGPR v -> m=v+8*lh, col n=lm.
#pragma unroll
    for (int mt = 0; mt < 2; ++mt)
#pragma unroll
        for (int nt = 0; nt < 4; ++nt) {
            int n = n0 + n0w + nt * 16 + lm;
            float bv_ = bias[n];
#pragma unroll
            for (int v = 0; v < 8; ++v) {
                int m = m0 + m0w + 16 * mt + v + 8 * lh;
                Y[(size_t)m * EMB + n] = (bf16_t)(acc[mt][nt][v] + bv_);
            }
        }
}

// ===========================================================================
// Kernel 2: per-head V column sums: Vsum[b,h,d] = sum_i Vh[i,d]  (f32).
// ===========================================================================
__global__ __launch_bounds__(64) void vsum_kernel(const bf16_t* __restrict__ Yv,
                                                  float* __restrict__ Vsum)
{
    const int head = blockIdx.x;               // 0..31 = b*16+h
    const int b = head >> 4, h = head & 15;
    const int d = threadIdx.x;                 // 0..63
    const bf16_t* base = Yv + (size_t)(b * SEQ + h * 128) * EMB;
    float s = 0.f;
    for (int rr = 0; rr < 128; ++rr)
#pragma unroll
        for (int g = 0; g < 16; ++g)
            s += (float)base[rr * EMB + g * DH + d];
    Vsum[head * DH + d] = s;
}

// ===========================================================================
// Kernel 3: windowed attention, single pass.
//   out[i,:] = ( sum_{win}(e^{s_ij}-1) V[j,:] + Vsum ) / (2048 + sum_{win}(e^{s_ij}-1))
// One WG = (head, 128-row block); 8 waves each own a 16-row tile.
// K window staged row-major via async global->LDS; V window staged transposed
// (manual). P tile bounces through per-wave LDS (C-layout -> A-layout).
// ===========================================================================
__global__ __launch_bounds__(256) void attn_kernel(
    const bf16_t* __restrict__ Yq, const bf16_t* __restrict__ Yk, const bf16_t* __restrict__ Yv,
    const float* __restrict__ Vsum, float* __restrict__ out)
{
    constexpr int JW = 416;                 // staged window width (covers jj < 400)
    __shared__ bf16_t Ks[JW][72];           // K[jj][d]        59904 B (row stride 144B)
    __shared__ bf16_t Vt[DH][JW + 8];       // V^T[d][jj]      54272 B
    __shared__ bf16_t Pw[8][16][40];        // per-wave P tile 10240 B

    const int head = blockIdx.x;            // 0..31
    const int b = head >> 4, h = head & 15;
    const int r0  = blockIdx.y * 128;       // virtual row base of this WG
    const int jlo = r0 - WSZ;               // staged window start (may be <0)
    const int tid = threadIdx.x, lane = tid & 31, wave = tid >> 5;
    const int lm = lane & 15, lh = lane >> 4;
    const int rowbase = b * SEQ + h * 128;  // Y row of this head's block

    // ---- stage K window (async, row-major) and V window (transposed) ----
    for (int idx = tid; idx < JW * 8; idx += 256) {   // 16B chunks: (jj, 8 d's)
        int jj = idx >> 3, c = (idx & 7) * 8;
        int j = jlo + jj;
        j = j < 0 ? 0 : (j > SEQ - 1 ? SEQ - 1 : j);  // clamp; masked later
        size_t off = (size_t)(rowbase + (j >> 4)) * EMB + (j & 15) * DH + c;
        async_copy16(Yk + off, &Ks[jj][c]);
        v8bf vv = *(const v8bf*)(Yv + off);
#pragma unroll
        for (int e = 0; e < 8; ++e) Vt[c + e][jj] = vv[e];
    }
    wait_async();
    __syncthreads();

    // ---- Q A-fragments (16x64 tile == one contiguous Yq row) ----
    const int i0 = r0 + 16 * wave;
    const bf16_t* qrow = Yq + (size_t)(rowbase + (i0 >> 4)) * EMB;
    v16bf aq[2];
#pragma unroll
    for (int kc = 0; kc < 2; ++kc) {
        const bf16_t* p = qrow + lm * DH + 32 * kc + 16 * lh;
        v16bf a;
#pragma unroll
        for (int e = 0; e < 8; ++e) { a[e] = p[e]; a[8 + e] = p[8 + e]; }
        aq[kc] = a;
    }

    v8f pv[4];
#pragma unroll
    for (int dt = 0; dt < 4; ++dt) pv[dt] = zero8();
    float dpart[8];
#pragma unroll
    for (int v = 0; v < 8; ++v) dpart[v] = 0.f;

    const float scale = 0.125f;             // D^-0.5 = 1/8

    // ---- 9 chunks of 32 columns: window [i0-128, i0+160) (over-cover masked) ----
    for (int t = 0; t < 9; ++t) {
        const int jj0 = 16 * wave + 32 * t;   // chunk start in staged coords
        const int j0  = jlo + jj0;            // absolute column start

        // two 16x16 score tiles -> e = exp(s/8)-1 (0 outside window) -> Pw
#pragma unroll
        for (int ht = 0; ht < 2; ++ht) {
            v8f sc = zero8();
#pragma unroll
            for (int kc = 0; kc < 2; ++kc) {
                const bf16_t* p = &Ks[jj0 + 16 * ht + lm][32 * kc + 16 * lh];
                v16bf bb;
#pragma unroll
                for (int e = 0; e < 16; ++e) bb[e] = p[e];
                sc = WMMA_BF16(aq[kc], bb, sc);
            }
            const int j = j0 + 16 * ht + lm;
#pragma unroll
            for (int v = 0; v < 8; ++v) {
                const int i = i0 + v + 8 * lh;
                const bool ok = (j >= 0) & (j < SEQ) & (j >= i - WSZ) & (j < i + WSZ);
                const float ev = ok ? (__expf(sc[v] * scale) - 1.0f) : 0.0f;
                dpart[v] += ev;
                Pw[wave][v + 8 * lh][16 * ht + lm] = (bf16_t)ev;
            }
        }

        // P A-fragment (K=32) and 4 PV WMMAs (one per 16-wide d tile)
        {
            const bf16_t* p = &Pw[wave][lm][16 * lh];
            v16bf ap;
#pragma unroll
            for (int e = 0; e < 8; ++e) { ap[e] = p[e]; ap[8 + e] = p[8 + e]; }
#pragma unroll
            for (int dt = 0; dt < 4; ++dt) {
                const bf16_t* pb = &Vt[16 * dt + lm][jj0 + 16 * lh];
                v16bf bb;
#pragma unroll
                for (int e = 0; e < 16; ++e) bb[e] = pb[e];
                pv[dt] = WMMA_BF16(ap, bb, pv[dt]);
            }
        }
    }

    // ---- row sums: reduce per-lane partials across the 16-lane half-group ----
#pragma unroll
    for (int v = 0; v < 8; ++v) {
        float s = dpart[v];
        s += __shfl_xor(s, 1, 32);
        s += __shfl_xor(s, 2, 32);
        s += __shfl_xor(s, 4, 32);
        s += __shfl_xor(s, 8, 32);
        dpart[v] = s;
    }

    // ---- epilogue: out[b, i, h*64 + d] = (pv + Vsum[d]) / (2048 + rowsum) ----
#pragma unroll
    for (int dt = 0; dt < 4; ++dt) {
        const float vs = Vsum[head * DH + 16 * dt + lm];
#pragma unroll
        for (int v = 0; v < 8; ++v) {
            const int i = i0 + v + 8 * lh;
            const float denom = (float)SEQ + dpart[v];
            out[(size_t)(b * SEQ + i) * EMB + h * DH + 16 * dt + lm] =
                (pv[dt][v] + vs) / denom;
        }
    }
}

// ===========================================================================
// Launch. Inputs (x, Wq, bq, Wk, bk, Wv, bv) f32; output f32 [2,2048,1024].
// Workspace: Yq/Yk/Yv bf16 (24 MB) + Vsum (8 KB) + xbf (8 MB) + Wbf (6 MB).
// ===========================================================================
extern "C" void kernel_launch(void* const* d_in, const int* in_sizes, int n_in,
                              void* d_out, int out_size, void* d_ws, size_t ws_size,
                              hipStream_t stream)
{
    (void)in_sizes; (void)n_in; (void)out_size; (void)ws_size;
    const float* x  = (const float*)d_in[0];
    const float* Wq = (const float*)d_in[1];
    const float* bq = (const float*)d_in[2];
    const float* Wk = (const float*)d_in[3];
    const float* bk = (const float*)d_in[4];
    const float* Wv = (const float*)d_in[5];
    const float* bv = (const float*)d_in[6];
    float* out = (float*)d_out;

    const size_t ybytes = (size_t)MTOT * EMB * sizeof(bf16_t);   // 8 MB
    const size_t wbytes = (size_t)EMB * EMB * sizeof(bf16_t);    // 2 MB
    char* ws = (char*)d_ws;
    bf16_t* Yq   = (bf16_t*)(ws);
    bf16_t* Yk   = (bf16_t*)(ws + ybytes);
    bf16_t* Yv   = (bf16_t*)(ws + 2 * ybytes);
    float*  Vsum = (float*)(ws + 3 * ybytes);
    bf16_t* xbf  = (bf16_t*)(ws + 3 * ybytes + 65536);
    bf16_t* Wqb  = (bf16_t*)(ws + 3 * ybytes + 65536 + ybytes);
    bf16_t* Wkb  = (bf16_t*)(ws + 3 * ybytes + 65536 + ybytes + wbytes);
    bf16_t* Wvb  = (bf16_t*)(ws + 3 * ybytes + 65536 + ybytes + 2 * wbytes);

    // f32 -> bf16 pre-convert (x and weights)
    {
        int n4x = MTOT * EMB / 4, n4w = EMB * EMB / 4;
        cvt_kernel<<<dim3((n4x + 255) / 256), 256, 0, stream>>>(x,  xbf, n4x);
        cvt_kernel<<<dim3((n4w + 255) / 256), 256, 0, stream>>>(Wq, Wqb, n4w);
        cvt_kernel<<<dim3((n4w + 255) / 256), 256, 0, stream>>>(Wk, Wkb, n4w);
        cvt_kernel<<<dim3((n4w + 255) / 256), 256, 0, stream>>>(Wv, Wvb, n4w);
    }

    qkv_gemm_kernel<<<dim3(MTOT / 128, EMB / 128, 3), 256, 0, stream>>>(
        xbf, Wqb, Wkb, Wvb, bq, bk, bv, Yq, Yk, Yv);
    vsum_kernel<<<dim3(NB * NH), 64, 0, stream>>>(Yv, Vsum);
    attn_kernel<<<dim3(NB * NH, SEQ / 128), 256, 0, stream>>>(Yq, Yk, Yv, Vsum, out);
}
